// GraphForward_39737037423445
// MI455X (gfx1250) — matrix operand order repaired and checked by reading, not compiled, not run on hardware
//
#include <hip/hip_runtime.h>
#include <math.h>

typedef __attribute__((ext_vector_type(16))) _Float16 v16h;
typedef __attribute__((ext_vector_type(8)))  _Float16 v8h;
typedef __attribute__((ext_vector_type(8)))  float    v8f;
typedef __attribute__((ext_vector_type(4)))  unsigned int v4u;
typedef __attribute__((ext_vector_type(8)))  int      v8i;
typedef __attribute__((ext_vector_type(4)))  int      v4i;

#define BTOT 32768

// ---- workspace layout (bytes) ----
#define WG1_OFF 0          // f16 [512][128]  gn1 {Wk,Wq,Wv,Ws}
#define WG2_OFF 131072     // f16 [256][128]  gn2 {Wk,Wq,Wv,Ws}
#define BG1_OFF 196608     // f32 [512]
#define BG2_OFF 198656     // f32 [256]
#define H_OFF   262144     // f16 [B*13][128]

// incoming-edge adjacency (dst -> list of src)
__constant__ int c_adjN[13] = {4, 2,2,2,2, 2,2,2,2, 1,1,1,1};
__constant__ int c_adjS[13][4] = {
    {1,2,3,4},
    {0,5,0,0},{0,6,0,0},{0,7,0,0},{0,8,0,0},
    {1,9,0,0},{2,10,0,0},{3,11,0,0},{4,12,0,0},
    {5,0,0,0},{6,0,0,0},{7,0,0,0},{8,0,0,0}};

static __device__ __forceinline__ float eluf(float x) {
    return x > 0.f ? x : (expf(x) - 1.f);
}
static __device__ __forceinline__ v16h mk16(v8h lo, v8h hi) {
    return __builtin_shufflevector(lo, hi, 0,1,2,3,4,5,6,7,8,9,10,11,12,13,14,15);
}

// ---------------- weight prep: fp32 -> packed f16 B-matrices ----------------
__global__ void __launch_bounds__(256) prep_kernel(
    const float* k1, const float* q1, const float* v1, const float* s1,
    const float* bk1, const float* bq1, const float* bv1, const float* bs1,
    const float* k2, const float* q2, const float* v2, const float* s2,
    const float* bk2, const float* bq2, const float* bv2, const float* bs2,
    _Float16* Wg1, float* bg1, _Float16* Wg2, float* bg2)
{
    int tid = blockIdx.x * 256 + threadIdx.x;
    if (tid < 65536) {                       // gn1 weights: [512][128]
        int n = tid >> 7, k = tid & 127;
        int g = n >> 7, nl = n & 127;
        const float* W = (g == 0) ? k1 : (g == 1) ? q1 : (g == 2) ? v1 : s1;
        Wg1[tid] = (_Float16)W[nl * 128 + k];
    } else if (tid < 98304) {                // gn2 weights: [256][128]
        int t2 = tid - 65536;
        int n = t2 >> 7, k = t2 & 127;
        int g = n >> 6, nl = n & 63;
        const float* W = (g == 0) ? k2 : (g == 1) ? q2 : (g == 2) ? v2 : s2;
        Wg2[t2] = (_Float16)W[nl * 128 + k];
    } else if (tid < 98816) {                // gn1 biases [512]
        int i = tid - 98304;
        int g = i >> 7, nl = i & 127;
        const float* b = (g == 0) ? bk1 : (g == 1) ? bq1 : (g == 2) ? bv1 : bs1;
        bg1[i] = b[nl];
    } else if (tid < 99072) {                // gn2 biases [256]
        int i = tid - 98816;
        int g = i >> 6, nl = i & 63;
        const float* b = (g == 0) ? bk2 : (g == 1) ? bq2 : (g == 2) ? bv2 : bs2;
        bg2[i] = b[nl];
    }
}

// ---------------- node encoders (scalar MLP, weights in LDS) ----------------
__global__ void __launch_bounds__(256) encoder_kernel(
    const float* obs, const float* act, const float* lat,
    const float* bW1, const float* bb1, const float* bW2, const float* bb2,
    const float* hW1, const float* hb1, const float* hW2, const float* hb2,
    const float* cW1, const float* cb1, const float* cW2, const float* cb2,
    _Float16* H)
{
    __shared__ float sW1[128 * 9];
    __shared__ float sb1[128];
    __shared__ float sW2[64 * 128];
    __shared__ float sb2[64];

    const int NB0 = BTOT / 256;              // 128 blocks: base nodes
    const int NB1 = NB0 + (8 * BTOT) / 256;  // +1024 blocks: hip/thigh
    int blk = blockIdx.x, tid = threadIdx.x;
    int region = (blk < NB0) ? 0 : ((blk < NB1) ? 1 : 2);

    const float *W1, *b1, *W2, *b2;
    int IN;
    if (region == 0)      { W1 = bW1; b1 = bb1; W2 = bW2; b2 = bb2; IN = 9; }
    else if (region == 1) { W1 = hW1; b1 = hb1; W2 = hW2; b2 = hb2; IN = 3; }
    else                  { W1 = cW1; b1 = cb1; W2 = cW2; b2 = cb2; IN = 3; }

    for (int i = tid; i < 128 * IN; i += 256) sW1[i] = W1[i];
    for (int i = tid; i < 128;      i += 256) sb1[i] = b1[i];
    for (int i = tid; i < 8192;     i += 256) sW2[i] = W2[i];
    for (int i = tid; i < 64;       i += 256) sb2[i] = b2[i];
    __syncthreads();

    int b, node;
    float in[9];
    if (region == 0) {
        b = blk * 256 + tid; node = 0;
        #pragma unroll
        for (int t = 0; t < 9; ++t) in[t] = obs[(size_t)b * 45 + t];
    } else if (region == 1) {
        int g = (blk - NB0) * 256 + tid;
        b = g >> 3; int j = g & 7;
        if (j < 4) {   // hip nodes 1..4
            node = 1 + j;
            in[0] = obs[(size_t)b * 45 + 9 + 3 * j];
            in[1] = obs[(size_t)b * 45 + 21 + 3 * j];
            in[2] = act[(size_t)b * 12 + 3 * j];
        } else {       // thigh nodes 5..8 (hip weights)
            int i2 = j - 4; node = 5 + i2;
            in[0] = obs[(size_t)b * 45 + 10 + 3 * i2];
            in[1] = obs[(size_t)b * 45 + 22 + 3 * i2];
            in[2] = act[(size_t)b * 12 + 3 * i2 + 1];
        }
    } else {
        int g = (blk - NB1) * 256 + tid;
        b = g >> 2; int i2 = g & 3; node = 9 + i2;
        in[0] = obs[(size_t)b * 45 + 11 + 3 * i2];
        in[1] = obs[(size_t)b * 45 + 23 + 3 * i2];
        in[2] = act[(size_t)b * 12 + 3 * i2 + 2];
    }

    float out[64];
    #pragma unroll
    for (int o = 0; o < 64; ++o) out[o] = sb2[o];
    for (int j = 0; j < 128; ++j) {
        float z = sb1[j];
        for (int t = 0; t < IN; ++t) z += in[t] * sW1[j * IN + t];
        float a = eluf(z);
        #pragma unroll 8
        for (int o = 0; o < 64; ++o) out[o] += a * sW2[o * 128 + j];
    }

    size_t ro = ((size_t)b * 13 + node) * 128;
    for (int o = 0; o < 64; ++o) H[ro + o] = (_Float16)out[o];
    const float* lr = lat + ((size_t)b * 13 + node) * 64;
    for (int d = 0; d < 64; ++d) H[ro + 64 + d] = (_Float16)lr[d];
}

// ------------- fused gated-graph-conv layer: WMMA GEMM + gating -------------
// WG = 16 batch elems = 208 rows = 13 M-tiles. A-tile + KQVS live in LDS.
// GS: per-matrix output width (gn1:128, gn2:64). nChunks: 64-wide d-chunks.
__global__ void __launch_bounds__(256) gn_kernel(
    _Float16* H, const _Float16* Wg, const float* bg,
    int GS, int nChunks, int doElu)
{
    extern __shared__ char smem[];
    const int SA = 136;                                   // padded A stride (f16)
    _Float16* At = (_Float16*)smem;                       // 208 x 136
    _Float16* Kq = (_Float16*)(smem + 208 * SA * 2);      // 208 x 256

    int tid = threadIdx.x;
    int lane = tid & 31, wv = tid >> 5;
    int half = lane >> 4, l16 = lane & 15;
    size_t rb0 = (size_t)blockIdx.x * 16 * 13;            // first global row

#if __has_builtin(__builtin_amdgcn_tensor_load_to_lds) && \
    __has_builtin(__builtin_amdgcn_s_wait_tensorcnt)
    // Tensor Data Mover: each wave DMAs its 26-row slice of the A tile.
    // data_size=2B, tile 128x26, pad 4 DWORDs after every 64 DWORDs (one
    // 256B row) -> LDS row stride 272B == SA f16, matching WMMA fragment
    // addressing below.
    {
        unsigned int ldsBase =
            (unsigned int)(uintptr_t)At + (unsigned int)(wv * 26) * (SA * 2);
        unsigned long long gbase =
            (unsigned long long)(uintptr_t)(H + (rb0 + (size_t)wv * 26) * 128);
        v4u g0;
        g0.x = 1u;                                    // count=1, user mode
        g0.y = ldsBase;                               // lds_addr
        g0.z = (unsigned int)gbase;                   // global_addr[31:0]
        g0.w = (unsigned int)((gbase >> 32) & 0x1FFFFFFu) | (2u << 30); // type=2
        v8i g1;
        g1[0] = (1 << 16)          // data_size = 2 bytes
              | (1 << 20)          // pad_enable
              | (5 << 22)          // pad_interval: 64 DWORDs (=256B row)
              | (3 << 25);         // pad_amount: 4 DWORDs (=16B)
        g1[1] = (128 << 16);       // tensor_dim0 = 128 elements
        g1[2] = (26 << 16);        // tensor_dim1 = 26 rows
        g1[3] = (128 << 16);       // tile_dim0 = 128
        g1[4] = 26;                // tile_dim1 = 26
        g1[5] = 128;               // tensor_dim0_stride = 128 elements
        g1[6] = 0;
        g1[7] = 0;
        v4i gz = {0, 0, 0, 0};
#if __has_include(<hip/amd_detail/amd_gfx1250_TDM.h>)
        v8i gz8 = {0, 0, 0, 0, 0, 0, 0, 0};
        __builtin_amdgcn_tensor_load_to_lds(g0, g1, gz, gz, gz8, 0);
#else
        __builtin_amdgcn_tensor_load_to_lds(g0, g1, gz, gz, 0);
#endif
        __builtin_amdgcn_s_wait_tensorcnt(0);
    }
#else
    // fallback: cooperative b128 copy (13 iters per thread)
    for (int t = tid; t < 208 * 16; t += 256) {
        int row = t >> 4, c = t & 15;
        *(v8h*)(At + row * SA + c * 8) =
            *(const v8h*)(H + (rb0 + row) * 128 + c * 8);
    }
#endif
    __syncthreads();

    for (int ch = 0; ch < nChunks; ++ch) {
        // col-slot outer (2 per wave), row-tile inner: B fragments + bias
        // stay in VGPRs across all 13 row tiles -> WMMA waits only on LDS.
        for (int cs = wv; cs < 16; cs += 8) {
            int group = cs >> 2, sub = cs & 3;            // k/q/v/s, 16-col sub
            int gc0 = group * GS + ch * 64 + sub * 16;    // col in weight mtx
            int lc0 = group * 64 + sub * 16;              // col in LDS KQVS
            // B: lane<16 holds K {kk..kk+7, kk+16..kk+23} of column gc0+l16
            const _Float16* bp = Wg + (size_t)(gc0 + l16) * 128 + half * 8;
            v16h bf0 = mk16(*(const v8h*)(bp +  0), *(const v8h*)(bp +  16));
            v16h bf1 = mk16(*(const v8h*)(bp + 32), *(const v8h*)(bp +  48));
            v16h bf2 = mk16(*(const v8h*)(bp + 64), *(const v8h*)(bp +  80));
            v16h bf3 = mk16(*(const v8h*)(bp + 96), *(const v8h*)(bp + 112));
            float bb = bg[gc0 + l16];

            for (int rt = 0; rt < 13; ++rt) {
                v8f acc = {bb, bb, bb, bb, bb, bb, bb, bb};   // bias in C
                // A: lane<16 holds K {kk..kk+7, kk+16..kk+23} of row l16
                const _Float16* ap = At + (rt * 16 + l16) * SA + half * 8;
                acc = __builtin_amdgcn_wmma_f32_16x16x32_f16(false,
                        mk16(*(const v8h*)(ap +  0), *(const v8h*)(ap + 16)),
                        false, bf0, (short)0, acc, false, false);
                acc = __builtin_amdgcn_wmma_f32_16x16x32_f16(false,
                        mk16(*(const v8h*)(ap + 32), *(const v8h*)(ap + 48)),
                        false, bf1, (short)0, acc, false, false);
                acc = __builtin_amdgcn_wmma_f32_16x16x32_f16(false,
                        mk16(*(const v8h*)(ap + 64), *(const v8h*)(ap + 80)),
                        false, bf2, (short)0, acc, false, false);
                acc = __builtin_amdgcn_wmma_f32_16x16x32_f16(false,
                        mk16(*(const v8h*)(ap + 96), *(const v8h*)(ap + 112)),
                        false, bf3, (short)0, acc, false, false);
                // C/D: vgpr r, lane<16 -> (M=r, N=l16); lane>=16 -> (M=8+r)
                #pragma unroll
                for (int r = 0; r < 8; ++r)
                    Kq[(rt * 16 + half * 8 + r) * 256 + lc0 + l16] =
                        (_Float16)acc[r];
            }
        }
        __syncthreads();

        // gating epilogue: out[dst] = s[dst] + sum_src sigmoid(k+q)*v, (elu)
        for (int idx = tid; idx < 208 * 64; idx += 256) {
            int row = idx >> 6, dl = idx & 63;
            int bl = row / 13, node = row - bl * 13, base = bl * 13;
            float kd = (float)Kq[row * 256 + dl];
            float sd = (float)Kq[row * 256 + 192 + dl];
            float m = 0.f;
            int cnt = c_adjN[node];
            for (int e = 0; e < cnt; ++e) {
                int sr = base + c_adjS[node][e];
                float qv = (float)Kq[sr * 256 + 64 + dl];
                float vv = (float)Kq[sr * 256 + 128 + dl];
                m += vv * (1.f / (1.f + expf(-(kd + qv))));
            }
            float o = sd + m;
            if (doElu) o = (o > 0.f) ? o : (expf(o) - 1.f);
            H[(rb0 + row) * 128 + ch * 64 + dl] = (_Float16)o;  // in-place
        }
        __syncthreads();
    }
}

// ---------------- decoders (scalar, weights + inputs staged in LDS) ---------
__global__ void __launch_bounds__(128) dec_kernel(
    const _Float16* H,
    const float* dbW1, const float* dbb1, const float* dbW2, const float* dbb2,
    const float* dlW1, const float* dlb1, const float* dlW2, const float* dlb2,
    float* out)
{
    extern __shared__ char smem[];
    int blk = blockIdx.x, tid = threadIdx.x;
    const int NBB = BTOT / 128;                // 256 base blocks, then 1024 leg

    if (blk < NBB) {
        // base decoder: 64 -> 128 -> 12
        _Float16* W1h = (_Float16*)smem;                   // 128*64
        _Float16* W2h = W1h + 128 * 64;                    // 12*128
        float* b1s = (float*)(W2h + 12 * 128);
        float* b2s = b1s + 128;
        _Float16* inT = (_Float16*)(b2s + 16);             // [128][80]
        for (int i = tid; i < 8192; i += 128) W1h[i] = (_Float16)dbW1[i];
        for (int i = tid; i < 1536; i += 128) W2h[i] = (_Float16)dbW2[i];
        if (tid < 128) b1s[tid] = dbb1[tid];
        if (tid < 12)  b2s[tid] = dbb2[tid];
        int b = blk * 128 + tid;
        const _Float16* hr = H + (size_t)b * 13 * 128;     // node 0, dims 0..63
        #pragma unroll
        for (int t8 = 0; t8 < 8; ++t8)
            *(v8h*)(inT + tid * 80 + t8 * 8) = *(const v8h*)(hr + t8 * 8);
        __syncthreads();

        float o[12];
        #pragma unroll
        for (int q = 0; q < 12; ++q) o[q] = b2s[q];
        for (int j = 0; j < 128; ++j) {
            float z = b1s[j];
            const v8h* wr = (const v8h*)(W1h + j * 64);
            const v8h* xr = (const v8h*)(inT + tid * 80);
            #pragma unroll
            for (int t8 = 0; t8 < 8; ++t8) {
                v8h w = wr[t8], x = xr[t8];
                #pragma unroll
                for (int u = 0; u < 8; ++u) z += (float)x[u] * (float)w[u];
            }
            float a = eluf(z);
            #pragma unroll
            for (int q = 0; q < 12; ++q) o[q] += a * (float)W2h[q * 128 + j];
        }
        float* orow = out + (size_t)b * 52;
        #pragma unroll
        for (int q = 0; q < 12; ++q) orow[q] = o[q];
    } else {
        // leg decoder: 256 -> 128 -> 10, thread = (batch, leg)
        _Float16* W1h = (_Float16*)smem;                   // 128*256
        _Float16* W2h = W1h + 128 * 256;                   // 10*128
        float* b1s = (float*)(W2h + 10 * 128);
        float* b2s = b1s + 128;
        _Float16* inT = (_Float16*)(b2s + 16);             // [128][264]
        for (int i = tid; i < 32768; i += 128) W1h[i] = (_Float16)dlW1[i];
        for (int i = tid; i < 1280;  i += 128) W2h[i] = (_Float16)dlW2[i];
        if (tid < 128) b1s[tid] = dlb1[tid];
        if (tid < 10)  b2s[tid] = dlb2[tid];

        int gid = (blk - NBB) * 128 + tid;
        int b = gid >> 2, leg = gid & 3;
        int nd[4] = {0, 1 + leg, 5 + leg, 9 + leg};        // FL/FR/RL/RR nodes
        #pragma unroll
        for (int j = 0; j < 4; ++j) {
            const _Float16* hr = H + ((size_t)b * 13 + nd[j]) * 128;
            #pragma unroll
            for (int t8 = 0; t8 < 8; ++t8)
                *(v8h*)(inT + tid * 264 + j * 64 + t8 * 8) =
                    *(const v8h*)(hr + t8 * 8);
        }
        __syncthreads();

        float o[10];
        #pragma unroll
        for (int q = 0; q < 10; ++q) o[q] = b2s[q];
        for (int j = 0; j < 128; ++j) {
            float z = b1s[j];
            const v8h* wr = (const v8h*)(W1h + j * 256);
            const v8h* xr = (const v8h*)(inT + tid * 264);
            #pragma unroll
            for (int t8 = 0; t8 < 32; ++t8) {
                v8h w = wr[t8], x = xr[t8];
                #pragma unroll
                for (int u = 0; u < 8; ++u) z += (float)x[u] * (float)w[u];
            }
            float a = eluf(z);
            #pragma unroll
            for (int q = 0; q < 10; ++q) o[q] += a * (float)W2h[q * 128 + j];
        }
        float* orow = out + (size_t)b * 52;
        orow[12 + 3 * leg + 0] = o[0];   // pos
        orow[12 + 3 * leg + 1] = o[1];
        orow[12 + 3 * leg + 2] = o[2];
        orow[24 + 3 * leg + 0] = o[3];   // vel
        orow[24 + 3 * leg + 1] = o[4];
        orow[24 + 3 * leg + 2] = o[5];
        orow[36 + 3 * leg + 0] = o[6];   // act
        orow[36 + 3 * leg + 1] = o[7];
        orow[36 + 3 * leg + 2] = o[8];
        orow[48 + leg] = o[9];           // contact
    }
}

extern "C" void kernel_launch(void* const* d_in, const int* in_sizes, int n_in,
                              void* d_out, int out_size, void* d_ws, size_t ws_size,
                              hipStream_t stream) {
    (void)in_sizes; (void)n_in; (void)out_size; (void)ws_size;
    const float* obs  = (const float*)d_in[0];
    const float* act  = (const float*)d_in[1];
    const float* lat  = (const float*)d_in[2];
    const float* bW1  = (const float*)d_in[3];
    const float* bb1  = (const float*)d_in[4];
    const float* bW2  = (const float*)d_in[5];
    const float* bb2  = (const float*)d_in[6];
    const float* hW1  = (const float*)d_in[7];
    const float* hb1  = (const float*)d_in[8];
    const float* hW2  = (const float*)d_in[9];
    const float* hb2  = (const float*)d_in[10];
    const float* cW1  = (const float*)d_in[11];
    const float* cb1  = (const float*)d_in[12];
    const float* cW2  = (const float*)d_in[13];
    const float* cb2  = (const float*)d_in[14];
    const float* g1k  = (const float*)d_in[15];
    const float* g1bk = (const float*)d_in[16];
    const float* g1q  = (const float*)d_in[17];
    const float* g1bq = (const float*)d_in[18];
    const float* g1v  = (const float*)d_in[19];
    const float* g1bv = (const float*)d_in[20];
    const float* g1s  = (const float*)d_in[21];
    const float* g1bs = (const float*)d_in[22];
    const float* g2k  = (const float*)d_in[23];
    const float* g2bk = (const float*)d_in[24];
    const float* g2q  = (const float*)d_in[25];
    const float* g2bq = (const float*)d_in[26];
    const float* g2v  = (const float*)d_in[27];
    const float* g2bv = (const float*)d_in[28];
    const float* g2s  = (const float*)d_in[29];
    const float* g2bs = (const float*)d_in[30];
    const float* dbW1 = (const float*)d_in[31];
    const float* dbb1 = (const float*)d_in[32];
    const float* dbW2 = (const float*)d_in[33];
    const float* dbb2 = (const float*)d_in[34];
    const float* dlW1 = (const float*)d_in[35];
    const float* dlb1 = (const float*)d_in[36];
    const float* dlW2 = (const float*)d_in[37];
    const float* dlb2 = (const float*)d_in[38];

    char* ws = (char*)d_ws;
    _Float16* Wg1 = (_Float16*)(ws + WG1_OFF);
    _Float16* Wg2 = (_Float16*)(ws + WG2_OFF);
    float*    bg1 = (float*)(ws + BG1_OFF);
    float*    bg2 = (float*)(ws + BG2_OFF);
    _Float16* H   = (_Float16*)(ws + H_OFF);
    float*    outp = (float*)d_out;

    prep_kernel<<<387, 256, 0, stream>>>(
        g1k, g1q, g1v, g1s, g1bk, g1bq, g1bv, g1bs,
        g2k, g2q, g2v, g2s, g2bk, g2bq, g2bv, g2bs,
        Wg1, bg1, Wg2, bg2);

    encoder_kernel<<<1664, 256, 0, stream>>>(
        obs, act, lat,
        bW1, bb1, bW2, bb2, hW1, hb1, hW2, hb2, cW1, cb1, cW2, cb2, H);

    const size_t gnSmem = 208 * 136 * sizeof(_Float16)   // A tile
                        + 208 * 256 * sizeof(_Float16);  // KQVS chunk
    gn_kernel<<<BTOT / 16, 256, gnSmem, stream>>>(H, Wg1, bg1, 128, 2, 1);
    gn_kernel<<<BTOT / 16, 256, gnSmem, stream>>>(H, Wg2, bg2, 64, 1, 0);

    const size_t decSmem = 128 * 256 * 2 + 10 * 128 * 2 + 512 + 64
                         + 128 * 264 * 2;                // leg branch (max)
    dec_kernel<<<BTOT / 128 + (4 * BTOT) / 128, 128, decSmem, stream>>>(
        H, dbW1, dbb1, dbW2, dbb2, dlW1, dlb1, dlW2, dlb2, outp);
}
